// Attention_1709396984084
// MI455X (gfx1250) — matrix operand order repaired and checked by reading, not compiled, not run on hardware
//
#include <hip/hip_runtime.h>

// ---------------------------------------------------------------------------
// Attention prefill for MI455X (gfx1250, wave32, WMMA + Tensor Data Mover).
//   X = x @ lin_w^T + b ; Q,K,V = X @ W_{q,k,v}
//   S = Q K^T (causal) ; P = softmax(S) ; F = P V
// GEMMs: TDM (tensor_load_to_lds) stages 128x64 bf16 tiles into LDS,
// double-buffered; waves consume via ds_load and v_wmma_f32_16x16x32_bf16.
// ---------------------------------------------------------------------------

typedef __bf16 bf16;
typedef unsigned int u32;
typedef unsigned long long u64;
typedef __attribute__((ext_vector_type(16))) __bf16 v16bf;
typedef __attribute__((ext_vector_type(8)))  __bf16 v8bf;
typedef __attribute__((ext_vector_type(8)))  float  v8f;
typedef __attribute__((ext_vector_type(4)))  u32    u32x4;
typedef __attribute__((ext_vector_type(8)))  u32    u32x8;

union V16U { v16bf v; v8bf h[2]; };

// ---------------------------------------------------------------------------
// TDM: issue tensor_load_to_lds of a rows x cols (bf16) 2D tile.
// D# packing per CDNA5 ISA ch.8 (group0: count/lds/global/type=2;
// group1: data_size=1(2B), tensor dims, tile dims, dim0 stride).
// The LDS destination pointer is both ptrtoint'd into the descriptor and
// passed as a register asm operand so the shared buffer is captured — the
// "memory" clobber then forces the compiler to treat the asm as writing the
// buffer (otherwise the ds_load fragment reads get folded to undef).
// ---------------------------------------------------------------------------
__device__ __forceinline__ void tdm_load_tile_bf16(
    bf16* lds_ptr, const bf16* gptr, long ld_elems,
    int tile_rows, int tile_cols, long tensor_cols)
{
  const u32 lds_off = (u32)(uintptr_t)lds_ptr;         // flat[31:0] = LDS offset
  const u64 ga = (u64)(uintptr_t)gptr;
  u32x4 g0;
  g0[0] = 1u;                                          // count=1, user D#
  g0[1] = lds_off;                                     // lds_addr
  g0[2] = (u32)(ga & 0xFFFFFFFFu);                     // global_addr[31:0]
  g0[3] = (u32)((ga >> 32) & 0x1FFFFFFu) | (2u << 30); // addr[56:32] | type=2

  const u64 td0 = (u64)tensor_cols;    // tensor_dim0 (elements per row)
  const u64 td1 = (u64)(1u << 30);     // tensor_dim1: large, tiles in-bounds
  const u64 s0  = (u64)ld_elems;       // tensor_dim0_stride (elements)
  u32x8 g1;
  g1[0] = (1u << 16);                                        // data_size=2B
  g1[1] = (u32)((td0 & 0xFFFFu) << 16);                      // dim0[15:0]
  g1[2] = (u32)((td0 >> 16) & 0xFFFFu) |
          (u32)((td1 & 0xFFFFu) << 16);                      // dim0[31:16]|dim1[15:0]
  g1[3] = (u32)((td1 >> 16) & 0xFFFFu) |
          ((u32)tile_cols << 16);                            // dim1[31:16]|tile_dim0
  g1[4] = (u32)tile_rows & 0xFFFFu;                          // tile_dim1 (tile_dim2=0)
  g1[5] = (u32)(s0 & 0xFFFFFFFFu);                           // dim0_stride[31:0]
  g1[6] = (u32)((s0 >> 32) & 0xFFFFu);                       // dim0_stride[47:32]
  g1[7] = 0u;                                                // dim1_stride (2D: unused)

  asm volatile("tensor_load_to_lds %0, %1"
               :: "s"(g0), "s"(g1), "v"(lds_ptr)
               : "memory");
}

// LDS A-fragment (16x32 bf16, MxK), tile row-major [rows][64]:
//   lanes 0-15 : M = lane,    K = 0..15
//   lanes 16-31: M = lane-16, K = 16..31
__device__ __forceinline__ v16bf lds_a_frag(const bf16* As, int m_local, int kk, int lane) {
  const bf16* p = As + (m_local + (lane & 15)) * 64 + kk + ((lane >> 4) << 4);
  V16U r;
  r.h[0] = *(const v8bf*)p;
  r.h[1] = *(const v8bf*)(p + 8);
  return r.v;
}

// LDS B-fragment (32x16 bf16, KxN) from N-major tile [rows=N][64]:
//   lanes 0-15 : N = lane,    K = 0..15 ; lanes 16-31: N = lane-16, K = 16..31
__device__ __forceinline__ v16bf lds_b_frag(const bf16* Bs, int n_local, int kk, int lane) {
  const bf16* p = Bs + (n_local + (lane & 15)) * 64 + kk + ((lane >> 4) << 4);
  return *(const v16bf*)p;
}

// ---------------------------------------------------------------------------
// Generic batched bf16 WMMA GEMM:  C[z] = A[z] * Bt[z]^T (+ bias)
// Block: 256 threads = 8 wave32 waves; block tile 128x128; wave tile 32x64.
// K-slab 64 staged via TDM into double-buffered LDS (4 x 16KB).
// Optional epilogues: fp32 store, bf16 store, bf16 transposed store.
// causal!=0: skip blocks entirely above the diagonal (n0 > m0+127).
// ---------------------------------------------------------------------------
__global__ __launch_bounds__(256) void attn_wmma_gemm_bf16(
    const bf16* __restrict__ A,  long lda,  long sA,
    const bf16* __restrict__ Bt, long ldb,  long sB,
    const float* __restrict__ bias,
    float* __restrict__ Cf,  long ldc,  long sCf,
    bf16*  __restrict__ Cb,  long ldcb, long sCb,
    bf16*  __restrict__ CbT, long ldct, long sCbT,
    int K, int causal)
{
  // LDS: A0 | A1 | B0 | B1, each 128x64 bf16 = 16KB (64KB total).
  // NOTE: selected via ternaries only — pointer arrays of LDS addresses become
  // constant-initialized globals with addrspacecast relocations that ld.lld
  // rejects ("unsupported expression in static initializer").
  __shared__ bf16 lds[4 * 128 * 64];

  const long m0 = (long)blockIdx.x * 128;
  const long n0 = (long)blockIdx.y * 128;
  if (causal && n0 > m0 + 127) return;           // fully-masked score block

  const long z = blockIdx.z;
  const bf16* Az  = A  + z * sA;
  const bf16* Btz = Bt + z * sB;

  const int lane = threadIdx.x & 31;
  const int wave = threadIdx.x >> 5;
  const int wm = (wave & 3) * 32;                 // wave M offset in block tile
  const int wn = (wave >> 2) * 64;                // wave N offset in block tile

  v8f acc[2][4] = {};

  const int nk = K >> 6;                          // K-slabs of 64 (K % 64 == 0)

  if (wave == 0) {                                // stage slab 0
    tdm_load_tile_bf16(lds,             Az  + m0 * lda, lda, 128, 64, K);
    tdm_load_tile_bf16(lds + 2 * 8192,  Btz + n0 * ldb, ldb, 128, 64, K);
  }

  for (int i = 0; i < nk; ++i) {
    if (wave == 0) {
      if (i + 1 < nk) {                           // stage slab i+1, other buffer
        const long k1 = (long)(i + 1) << 6;
        bf16* An = ((i + 1) & 1) ? (lds + 8192)     : lds;
        bf16* Bn = ((i + 1) & 1) ? (lds + 3 * 8192) : (lds + 2 * 8192);
        tdm_load_tile_bf16(An, Az  + m0 * lda + k1, lda, 128, 64, K);
        tdm_load_tile_bf16(Bn, Btz + n0 * ldb + k1, ldb, 128, 64, K);
        __builtin_amdgcn_s_wait_tensorcnt(2);     // slab i landed in LDS
      } else {
        __builtin_amdgcn_s_wait_tensorcnt(0);
      }
    }
    __syncthreads();                              // slab i visible to all waves

    const bf16* As = (i & 1) ? (lds + 8192)     : lds;
    const bf16* Bs = (i & 1) ? (lds + 3 * 8192) : (lds + 2 * 8192);

#pragma unroll
    for (int kk = 0; kk < 64; kk += 32) {
      v16bf a[2], b[4];
#pragma unroll
      for (int ii = 0; ii < 2; ++ii) a[ii] = lds_a_frag(As, wm + 16 * ii, kk, lane);
#pragma unroll
      for (int jj = 0; jj < 4; ++jj) b[jj] = lds_b_frag(Bs, wn + 16 * jj, kk, lane);
#pragma unroll
      for (int ii = 0; ii < 2; ++ii)
#pragma unroll
        for (int jj = 0; jj < 4; ++jj)
          acc[ii][jj] = __builtin_amdgcn_wmma_f32_16x16x32_bf16(
              false, a[ii], false, b[jj], (short)0, acc[ii][jj], false, false);
    }
    __syncthreads();                              // done reading slab i buffers
  }

  // Epilogue. C/D layout: VGPR r -> M = r + 8*(lane>=16), N = lane&15.
#pragma unroll
  for (int ii = 0; ii < 2; ++ii) {
#pragma unroll
    for (int jj = 0; jj < 4; ++jj) {
      const long n  = n0 + wn + 16 * jj + (lane & 15);
      const long mb = m0 + wm + 16 * ii + ((long)(lane >> 4) << 3);
      const float bv = bias ? bias[n] : 0.0f;
#pragma unroll
      for (int r = 0; r < 8; ++r) {
        const float v = acc[ii][jj][r] + bv;
        const long m = mb + r;
        if (Cf)  Cf [z * sCf  + m * ldc  + n] = v;
        if (Cb)  Cb [z * sCb  + m * ldcb + n] = (bf16)v;
        if (CbT) CbT[z * sCbT + n * ldct + m] = (bf16)v;
      }
    }
  }
}

// ---------------------------------------------------------------------------
// Causal row softmax: one 256-thread block per row of S (2048 wide).
// P[q][k] = exp(S[q][k]-max)/sum for k<=q, else 0 (PV GEMM reads dense K).
// ---------------------------------------------------------------------------
__global__ __launch_bounds__(256) void attn_softmax_causal(
    const float* __restrict__ Sm, bf16* __restrict__ P, int n /* = 2048 */)
{
  __shared__ float red[256];
  const long row = blockIdx.x;            // b*S + q
  const int  q   = (int)(row & (n - 1));  // n is a power of two
  const float* srow = Sm + row * (long)n;
  bf16*        prow = P  + row * (long)n;
  const int t = threadIdx.x;

  float vals[8];
  float m = -3.4e38f;
#pragma unroll
  for (int i = 0; i < 8; ++i) {
    const int k = t + i * 256;
    vals[i] = (k <= q) ? srow[k] : -3.4e38f;
    m = fmaxf(m, vals[i]);
  }
  red[t] = m; __syncthreads();
  for (int s = 128; s > 0; s >>= 1) {
    if (t < s) red[t] = fmaxf(red[t], red[t + s]);
    __syncthreads();
  }
  m = red[0]; __syncthreads();

  float e[8];
  float sum = 0.0f;
#pragma unroll
  for (int i = 0; i < 8; ++i) {
    const int k = t + i * 256;
    e[i] = (k <= q) ? __expf(vals[i] - m) : 0.0f;
    sum += e[i];
  }
  red[t] = sum; __syncthreads();
  for (int s = 128; s > 0; s >>= 1) {
    if (t < s) red[t] += red[t + s];
    __syncthreads();
  }
  const float inv = 1.0f / red[0];
#pragma unroll
  for (int i = 0; i < 8; ++i) prow[t + i * 256] = (bf16)(e[i] * inv);
}

// ---------------------------------------------------------------------------
// fp32 -> bf16 elementwise convert, and converting 1024x1024 transpose
// (weights stored N-major so WMMA B tiles are row-contiguous).
// ---------------------------------------------------------------------------
__global__ __launch_bounds__(256) void attn_cvt_bf16(
    const float* __restrict__ in, bf16* __restrict__ out, long n)
{
  const long i = (long)blockIdx.x * 256 + threadIdx.x;
  if (i < n) out[i] = (bf16)in[i];
}

__global__ __launch_bounds__(256) void attn_cvt_transpose_bf16(
    const float* __restrict__ in /*[1024][1024] d-major*/,
    bf16* __restrict__ out       /*[1024][1024] h-major*/)
{
  const long i = (long)blockIdx.x * 256 + threadIdx.x;  // over 1M elems
  const long d = i >> 10;
  const long h = i & 1023;
  out[h * 1024 + d] = (bf16)in[i];
}

// ---------------------------------------------------------------------------
// Host-side orchestration.
// d_out layout: F [8,2048,1024] | Kcache [8,2048,1024] | Vcache [8,2048,1024]
// Workspace (peak ~288 MB, temporally reused):
//   [0,32)MB Qbf | [32,64) Kbf | [64,96) Vt (H x B*S) |
//   [96,128) xbf | [128,160) Xbf | [160,168) bf16 weights   (all dead after QKV)
//   [96,224) S fp32 (reuses dead region) | [224,288) P bf16
// ---------------------------------------------------------------------------
extern "C" void kernel_launch(void* const* d_in, const int* in_sizes, int n_in,
                              void* d_out, int out_size, void* d_ws, size_t ws_size,
                              hipStream_t stream) {
  (void)in_sizes; (void)n_in; (void)out_size; (void)ws_size;
  const float* x  = (const float*)d_in[0];
  const float* lw = (const float*)d_in[1];
  const float* lb = (const float*)d_in[2];
  const float* wq = (const float*)d_in[3];
  const float* wk = (const float*)d_in[4];
  const float* wv = (const float*)d_in[5];
  float* out = (float*)d_out;

  const long B = 8, S = 2048, D = 1024, H = 1024;
  const long MS = B * S;                       // 16384 total rows

  char* w = (char*)d_ws;
  const size_t MBy = 1024ull * 1024ull;
  bf16* Qbf  = (bf16*)(w +   0 * MBy);
  bf16* Kbf  = (bf16*)(w +  32 * MBy);
  bf16* Vt   = (bf16*)(w +  64 * MBy);         // [H][B*S]
  bf16* xbf  = (bf16*)(w +  96 * MBy);
  bf16* Xbf  = (bf16*)(w + 128 * MBy);
  bf16* lwbf = (bf16*)(w + 160 * MBy);         // lin_w is (e,d) == Bt layout already
  bf16* wqT  = (bf16*)(w + 162 * MBy);
  bf16* wkT  = (bf16*)(w + 164 * MBy);
  bf16* wvT  = (bf16*)(w + 166 * MBy);
  float* Sm  = (float*)(w +  96 * MBy);        // reuses dead xbf/Xbf/weights
  bf16* Pbf  = (bf16*)(w + 224 * MBy);

  float* Fout   = out;
  float* Kcache = out + MS * H;
  float* Vcache = out + 2 * MS * H;

  // ---- precision conversion -------------------------------------------------
  attn_cvt_bf16<<<(unsigned)((MS * D) / 256), 256, 0, stream>>>(x, xbf, MS * D);
  attn_cvt_bf16<<<(unsigned)((D * D) / 256), 256, 0, stream>>>(lw, lwbf, D * D);
  attn_cvt_transpose_bf16<<<(unsigned)((D * H) / 256), 256, 0, stream>>>(wq, wqT);
  attn_cvt_transpose_bf16<<<(unsigned)((D * H) / 256), 256, 0, stream>>>(wk, wkT);
  attn_cvt_transpose_bf16<<<(unsigned)((D * H) / 256), 256, 0, stream>>>(wv, wvT);

  const dim3 blk(256);

  // ---- X = x @ lin_w^T + b  (M=16384, N=1024, K=1024) ----------------------
  attn_wmma_gemm_bf16<<<dim3((unsigned)(MS / 128), (unsigned)(D / 128), 1), blk, 0, stream>>>(
      xbf, D, 0, lwbf, D, 0, lb,
      nullptr, 0, 0, Xbf, D, 0, nullptr, 0, 0, (int)D, 0);

  // ---- Q = X @ Wq -----------------------------------------------------------
  attn_wmma_gemm_bf16<<<dim3((unsigned)(MS / 128), (unsigned)(H / 128), 1), blk, 0, stream>>>(
      Xbf, D, 0, wqT, D, 0, nullptr,
      nullptr, 0, 0, Qbf, H, 0, nullptr, 0, 0, (int)D, 0);

  // ---- K = X @ Wk  (bf16 for attention + fp32 KV-cache) ---------------------
  attn_wmma_gemm_bf16<<<dim3((unsigned)(MS / 128), (unsigned)(H / 128), 1), blk, 0, stream>>>(
      Xbf, D, 0, wkT, D, 0, nullptr,
      Kcache, H, 0, Kbf, H, 0, nullptr, 0, 0, (int)D, 0);

  // ---- V = X @ Wv  (fp32 cache + bf16 transposed [H][B*S] for PV GEMM) ------
  attn_wmma_gemm_bf16<<<dim3((unsigned)(MS / 128), (unsigned)(H / 128), 1), blk, 0, stream>>>(
      Xbf, D, 0, wvT, D, 0, nullptr,
      Vcache, H, 0, nullptr, 0, 0, Vt, MS, 0, (int)D, 0);

  // ---- S = Q @ K^T  per batch, causal block skip  (M=N=2048, K=1024) --------
  // Kbf is [S][H] row-major per batch, which is exactly the Bt layout needed.
  attn_wmma_gemm_bf16<<<dim3((unsigned)(S / 128), (unsigned)(S / 128), (unsigned)B), blk, 0, stream>>>(
      Qbf, H, S * H, Kbf, H, S * H, nullptr,
      Sm, S, S * S, nullptr, 0, 0, nullptr, 0, 0, (int)H, 1);

  // ---- P = softmax(S) causal ------------------------------------------------
  attn_softmax_causal<<<(unsigned)MS, blk, 0, stream>>>(Sm, Pbf, (int)S);

  // ---- F = P @ V  (M=2048, N=1024, K=2048), Bt = Vt columns per batch -------
  attn_wmma_gemm_bf16<<<dim3((unsigned)(S / 128), (unsigned)(H / 128), (unsigned)B), blk, 0, stream>>>(
      Pbf, S, S * S, Vt, MS, S, nullptr,
      Fout, H, S * H, nullptr, 0, 0, nullptr, 0, 0, (int)S, 0);
}
